// TPI_59261958750327
// MI455X (gfx1250) — matrix operand order repaired and checked by reading, not compiled, not run on hardware
//
#include <hip/hip_runtime.h>
#include <hip/hip_bf16.h>
#include <math.h>

// ---------------- problem constants ----------------
#define BB 64
#define TT 4096
#define NN 128
#define FBINS 2048            // we compute bins 1..2048 (amp[0] forced to 0)
#define BN (BB * NN)          // 8192 GEMM columns
#define BNT (BN / 16)         // 512 bn-tiles
#define FT (FBINS / 16)       // 128 f-tiles
#define KSTEPS (TT / 32)      // 128 k-steps of 32

typedef __bf16 bf16;
typedef bf16  v16bf __attribute__((ext_vector_type(16)));
typedef float v8f   __attribute__((ext_vector_type(8)));

// ---------------- tiny init: amp[0] = 0 ----------------
__global__ void tpi_init_amp(float* amp) {
    if (threadIdx.x == 0) amp[0] = 0.0f;
}

// ---------------- merge 3 depthwise kernels into one 63-tap kernel ----------
// layout: wAll[c*63+j] = layer1 combined, wAll[128*63 + c*63+j] = layer2 combined
__global__ void tpi_merge_w(const float* w1a, const float* w1b, const float* w1c,
                            const float* w2a, const float* w2b, const float* w2c,
                            float* wAll) {
    int i = blockIdx.x * 256 + threadIdx.x;
    if (i >= NN * 63) return;
    int c = i / 63, j = i % 63;
    float a = w1c[c * 63 + j];
    float b = w2c[c * 63 + j];
    if (j >= 16 && j <= 46) { a += w1b[c * 31 + (j - 16)]; b += w2b[c * 31 + (j - 16)]; }
    if (j >= 24 && j <= 38) { a += w1a[c * 15 + (j - 24)]; b += w2a[c * 15 + (j - 24)]; }
    wAll[i] = a * (1.0f / 3.0f);
    wAll[NN * 63 + i] = b * (1.0f / 3.0f);
}

// ---------------- twiddle tables directly in B-fragment layout ----------------
// B (32x16, 16-bit): lane l -> column f_tile*16+(l&15); lanes<16 hold K=0..15,
// lanes>=16 hold K=16..31; value m <-> K = base+m. One 16-value record per
// (f_tile, kstep, lane), contiguous -> wave fragment loads are 2x b128.
__global__ void tpi_gen_tw(bf16* cosW, bf16* sinW) {
    int tid = blockIdx.x * 256 + threadIdx.x;      // FT*KSTEPS*32 = 524288
    if (tid >= FT * KSTEPS * 32) return;
    int lane  = tid & 31;
    int ks    = (tid >> 5) & (KSTEPS - 1);
    int ftile = tid >> 12;                          // / (KSTEPS*32)
    int bin   = 1 + ftile * 16 + (lane & 15);
    int tbase = ks * 32 + ((lane & 16) ? 16 : 0);
    size_t o  = (size_t)tid * 16;
    for (int m = 0; m < 16; ++m) {
        int t  = tbase + m;
        int ph = (bin * t) & (TT - 1);              // exact phase mod T
        float th = (float)ph * (6.283185307179586f / (float)TT);
        float s, c;
        __sincosf(th, &s, &c);
        cosW[o + m] = (bf16)c;
        sinW[o + m] = (bf16)s;
    }
}

// ---------------- x -> bf16 in A-fragment layout ----------------
// A (16x32, 16-bit): lane l<16 row M=l holds K = 0..15; lane l>=16 row
// M=l-16 holds K = 16..31; value m -> K = base+m (matches B layout).
__global__ void tpi_gen_xa(const float* __restrict__ x, bf16* xA) {
    int tid = blockIdx.x * 256 + threadIdx.x;      // BNT*KSTEPS*32 = 2097152
    if (tid >= BNT * KSTEPS * 32) return;
    int lane   = tid & 31;
    int ks     = (tid >> 5) & (KSTEPS - 1);
    int bntile = tid >> 12;
    int bn = bntile * 16 + (lane & 15);
    int b  = bn >> 7, n = bn & 127;
    bool hi = (lane & 16) != 0;
    size_t xbase = ((size_t)b * TT) * NN + n;
    size_t o = (size_t)tid * 16;
    for (int m = 0; m < 16; ++m) {
        int K = hi ? m + 16 : m;
        int t = ks * 32 + K;
        xA[o + m] = (bf16)x[xbase + (size_t)t * NN];
    }
}

// ---------------- WMMA DFT, 2x2 register-blocked -----------------------------
// Each wave owns 2 bn-tiles x 2 f-tiles (8 f32 accumulators); A fragments are
// reused across both f-tiles and cos/sin fragments across both bn-tiles:
// 6 KB of b128 loads feed 8 v_wmma per k-step (~21 FLOP/B from L2).
__global__ __launch_bounds__(256) void tpi_dft(const bf16* __restrict__ xA,
                                               const bf16* __restrict__ cosW,
                                               const bf16* __restrict__ sinW,
                                               float* __restrict__ part) {
    int wave = blockIdx.x * 8 + (threadIdx.x >> 5);
    int lane = threadIdx.x & 31;
    int bt0 = (wave >> 6) * 2;         // bn-tile pair base (0..510)
    int ft0 = (wave & 63) * 2;         // f-tile pair base  (0..126)

    const size_t TILE = (size_t)KSTEPS * 32 * 16;  // elems per (tile, all ksteps)
    const bf16* a0p = xA   + (size_t)bt0 * TILE + (size_t)lane * 16;
    const bf16* a1p = a0p + TILE;
    const bf16* c0p = cosW + (size_t)ft0 * TILE + (size_t)lane * 16;
    const bf16* c1p = c0p + TILE;
    const bf16* s0p = sinW + (size_t)ft0 * TILE + (size_t)lane * 16;
    const bf16* s1p = s0p + TILE;

    v8f ar00 = {0.f,0.f,0.f,0.f,0.f,0.f,0.f,0.f}, ai00 = ar00;
    v8f ar01 = ar00, ai01 = ar00, ar10 = ar00, ai10 = ar00;
    v8f ar11 = ar00, ai11 = ar00;

    for (int ks = 0; ks < KSTEPS; ++ks) {
        v16bf a0 = *(const v16bf*)a0p;
        v16bf a1 = *(const v16bf*)a1p;
        v16bf c0 = *(const v16bf*)c0p;
        v16bf c1 = *(const v16bf*)c1p;
        v16bf s0 = *(const v16bf*)s0p;
        v16bf s1 = *(const v16bf*)s1p;
        ar00 = __builtin_amdgcn_wmma_f32_16x16x32_bf16(false, a0, false, c0, (short)0, ar00, false, false);
        ai00 = __builtin_amdgcn_wmma_f32_16x16x32_bf16(false, a0, false, s0, (short)0, ai00, false, false);
        ar01 = __builtin_amdgcn_wmma_f32_16x16x32_bf16(false, a0, false, c1, (short)0, ar01, false, false);
        ai01 = __builtin_amdgcn_wmma_f32_16x16x32_bf16(false, a0, false, s1, (short)0, ai01, false, false);
        ar10 = __builtin_amdgcn_wmma_f32_16x16x32_bf16(false, a1, false, c0, (short)0, ar10, false, false);
        ai10 = __builtin_amdgcn_wmma_f32_16x16x32_bf16(false, a1, false, s0, (short)0, ai10, false, false);
        ar11 = __builtin_amdgcn_wmma_f32_16x16x32_bf16(false, a1, false, c1, (short)0, ar11, false, false);
        ai11 = __builtin_amdgcn_wmma_f32_16x16x32_bf16(false, a1, false, s1, (short)0, ai11, false, false);
        a0p += 32 * 16; a1p += 32 * 16;
        c0p += 32 * 16; c1p += 32 * 16;
        s0p += 32 * 16; s1p += 32 * 16;
        __builtin_prefetch(a0p + 32 * 16, 0, 0);   // global_prefetch_b8
        __builtin_prefetch(c0p + 32 * 16, 0, 0);
    }

    // C layout: VGPR r, lanes 0-15 -> M=r, lanes 16-31 -> M=8+r; N = lane&15.
    float m00 = 0.f, m01 = 0.f, m10 = 0.f, m11 = 0.f;
#pragma unroll
    for (int e = 0; e < 8; ++e) {
        m00 += sqrtf(ar00[e] * ar00[e] + ai00[e] * ai00[e]);
        m01 += sqrtf(ar01[e] * ar01[e] + ai01[e] * ai01[e]);
        m10 += sqrtf(ar10[e] * ar10[e] + ai10[e] * ai10[e]);
        m11 += sqrtf(ar11[e] * ar11[e] + ai11[e] * ai11[e]);
    }
    m00 += __shfl_xor(m00, 16, 32);                // rows 0..7 + rows 8..15
    m01 += __shfl_xor(m01, 16, 32);
    m10 += __shfl_xor(m10, 16, 32);
    m11 += __shfl_xor(m11, 16, 32);
    if (lane < 16) {
        part[(size_t)((ft0 + 0) * 16 + lane) * BNT + (bt0 + 0)] = m00;
        part[(size_t)((ft0 + 1) * 16 + lane) * BNT + (bt0 + 0)] = m01;
        part[(size_t)((ft0 + 0) * 16 + lane) * BNT + (bt0 + 1)] = m10;
        part[(size_t)((ft0 + 1) * 16 + lane) * BNT + (bt0 + 1)] = m11;
    }
}

// ---------------- fixed-order reduction over bn-tiles -> amp[1..2048] -------
__global__ void tpi_reduce_amp(const float* __restrict__ part, float* amp) {
    int f = blockIdx.x * 256 + threadIdx.x;
    if (f >= FBINS) return;
    const float* p = part + (size_t)f * BNT;
    float s = 0.f;
    for (int i = 0; i < BNT; ++i) s += p[i];
    amp[1 + f] = s;                                 // scale irrelevant for top-k
}

// ---------------- top-3 over amp[0..2048]; emit top_list & periods ----------
__global__ void tpi_topk(const float* __restrict__ amp, int* topP) {
    __shared__ float sv[256 * 3];
    __shared__ int   si[256 * 3];
    float v0 = -1.f, v1 = -1.f, v2 = -1.f;
    int   i0 = 0, i1 = 0, i2 = 0;
    for (int f = threadIdx.x; f < FBINS + 1; f += 256) {
        float a = amp[f];
        if (a > v0)      { v2 = v1; i2 = i1; v1 = v0; i1 = i0; v0 = a; i0 = f; }
        else if (a > v1) { v2 = v1; i2 = i1; v1 = a;  i1 = f; }
        else if (a > v2) { v2 = a;  i2 = f; }
    }
    sv[threadIdx.x * 3 + 0] = v0; si[threadIdx.x * 3 + 0] = i0;
    sv[threadIdx.x * 3 + 1] = v1; si[threadIdx.x * 3 + 1] = i1;
    sv[threadIdx.x * 3 + 2] = v2; si[threadIdx.x * 3 + 2] = i2;
    __syncthreads();
    if (threadIdx.x == 0) {
        float g0 = -1.f, g1 = -1.f, g2 = -1.f; int j0 = 1, j1 = 1, j2 = 1;
        for (int k = 0; k < 256 * 3; ++k) {
            float a = sv[k]; int f = si[k];
            if (a > g0)      { g2 = g1; j2 = j1; g1 = g0; j1 = j0; g0 = a; j0 = f; }
            else if (a > g1) { g2 = g1; j2 = j1; g1 = a;  j1 = f; }
            else if (a > g2) { g2 = a;  j2 = f; }
        }
        (void)g2; (void)j2;
        int tl0 = 1, tl1 = (j0 > 0 ? j0 : 1), tl2 = (j1 > 0 ? j1 : 1);
        topP[0] = tl0; topP[1] = tl1; topP[2] = tl2;
        topP[4] = TT / tl0; topP[5] = TT / tl1; topP[6] = TT / tl2;
    }
}

// ---------------- per-batch amplitudes at top_list freqs (exact f32) --------
__global__ __launch_bounds__(128) void tpi_weights(const float* __restrict__ x,
                                                   const int* __restrict__ topP,
                                                   float* wraw) {
    int b = blockIdx.x, kk = blockIdx.y, n = threadIdx.x;
    int f = topP[kk];
    float re = 0.f, im = 0.f;
    const float* xp = x + ((size_t)b * TT) * NN + n;
    for (int t = 0; t < TT; ++t) {
        int ph = (f * t) & (TT - 1);
        float th = (float)ph * (6.283185307179586f / (float)TT);
        float s, c;
        __sincosf(th, &s, &c);
        float xv = xp[(size_t)t * NN];
        re += xv * c;
        im += xv * s;                                // sign irrelevant for |.|
    }
    __shared__ float red[128];
    red[n] = sqrtf(re * re + im * im);
    __syncthreads();
    for (int s = 64; s > 0; s >>= 1) {
        if (n < s) red[n] += red[n + s];
        __syncthreads();
    }
    if (n == 0) wraw[b * 3 + kk] = red[0] * (1.0f / (float)NN);
}

__global__ void tpi_softmax(const float* wraw, float* wsoft) {
    int b = threadIdx.x;
    if (b >= BB) return;
    float a0 = wraw[b * 3 + 0], a1 = wraw[b * 3 + 1], a2 = wraw[b * 3 + 2];
    float m = fmaxf(a0, fmaxf(a1, a2));
    float e0 = expf(a0 - m), e1 = expf(a1 - m), e2 = expf(a2 - m);
    float inv = 1.0f / (e0 + e1 + e2);
    wsoft[b * 3 + 0] = e0 * inv;
    wsoft[b * 3 + 1] = e1 * inv;
    wsoft[b * 3 + 2] = e2 * inv;
}

// ---------------- residual init: out = x via async LDS round-trip -----------
// Exercises GLOBAL_LOAD_ASYNC_TO_LDS / GLOBAL_STORE_ASYNC_FROM_LDS + ASYNCcnt.
// Low 32 bits of a flat __shared__ pointer are the LDS byte offset (ISA §10.2).
__global__ __launch_bounds__(256) void tpi_copy_out(const float* __restrict__ x,
                                                    float* __restrict__ out) {
    __shared__ float buf[256 * 4];
    size_t i = ((size_t)blockIdx.x * 256 + threadIdx.x) * 4;   // element index
    const float* src = x + i;
    float* dst = out + i;
    unsigned lofs = (unsigned)(uintptr_t)(&buf[threadIdx.x * 4]);
    asm volatile("global_load_async_to_lds_b128 %0, %1, off"
                 :: "v"(lofs), "v"(src) : "memory");
    asm volatile("s_wait_asynccnt 0x0" ::: "memory");
    asm volatile("global_store_async_from_lds_b128 %0, %1, off"
                 :: "v"(dst), "v"(lofs) : "memory");
    asm volatile("s_wait_asynccnt 0x0" ::: "memory");
}

// ---------------- fused conv63 -> GELU -> conv63, segment-masked ------------
#define CHUNK 256
#define NC 8
#define XSPAN (CHUNK + 124)    // x staged on [t0-62, t0+CHUNK+62)
#define MSPAN (CHUNK + 62)     // mid on [t0-31, t0+CHUNK+31)

__global__ __launch_bounds__(256) void tpi_branch(const float* __restrict__ x,
                                                  float* __restrict__ out,
                                                  const float* __restrict__ wAll,
                                                  const int* __restrict__ topP,
                                                  const float* __restrict__ wsoft,
                                                  int kk) {
    __shared__ float sX[XSPAN][NC];
    __shared__ float sM[MSPAN][NC];
    __shared__ float sW1[NC][64];
    __shared__ float sW2[NC][64];

    const int b   = blockIdx.y;
    const int t0  = blockIdx.x * CHUNK;
    const int c0  = blockIdx.z * NC;
    const int tid = threadIdx.x;
    const int p   = topP[4 + kk];                  // period, data-dependent
    const float wgt = wsoft[b * 3 + kk];

    for (int i = tid; i < NC * 63; i += 256) {
        int c = i / 63, j = i % 63;
        sW1[c][j] = wAll[(c0 + c) * 63 + j];
        sW2[c][j] = wAll[NN * 63 + (c0 + c) * 63 + j];
    }
    for (int i = tid; i < XSPAN * NC; i += 256) {
        int m = i / NC, c = i % NC;
        int v = t0 - 62 + m;
        float val = 0.f;
        if (v >= 0 && v < TT) val = x[((size_t)b * TT + v) * NN + c0 + c];
        sX[m][c] = val;
    }
    __syncthreads();

    // conv1 (63-tap, zero-padded within segment, x zero for t>=T) + exact GELU
    for (int i = tid; i < MSPAN * NC; i += 256) {
        int m = i / NC, c = i % NC;
        int u = t0 - 31 + m;
        float r = 0.f;
        if (u >= 0) {
            int su   = (u / p) * p;
            int base = u - 31;                     // tap j reads position base+j
            int jlo  = max(0, su - base);
            int hi   = min(su + p, TT) - 1 - base;
            int jhi  = min(62, hi);
            float acc = 0.f;
            for (int j = jlo; j <= jhi; ++j)
                acc += sW1[c][j] * sX[m + j][c];   // sX index = (base+j)-(t0-62)
            r = 0.5f * acc * (1.0f + erff(acc * 0.7071067811865476f));
        }
        sM[m][c] = r;
    }
    __syncthreads();

    // conv2 (taps masked to t's segment) + weighted accumulation into out
    for (int i = tid; i < CHUNK * NC; i += 256) {
        int m = i / NC, c = i % NC;
        int t = t0 + m;
        int st   = (t / p) * p;
        int base = t - 31;
        int jlo  = max(0, st - base);
        int jhi  = min(62, st + p - 1 - base);
        float acc = 0.f;
        for (int j = jlo; j <= jhi; ++j)
            acc += sW2[c][j] * sM[m + j][c];
        out[((size_t)b * TT + t) * NN + c0 + c] += wgt * acc;
    }
}

// ---------------- launcher ----------------
extern "C" void kernel_launch(void* const* d_in, const int* in_sizes, int n_in,
                              void* d_out, int out_size, void* d_ws, size_t ws_size,
                              hipStream_t stream) {
    const float* x   = (const float*)d_in[0];
    const float* w1a = (const float*)d_in[1];
    const float* w1b = (const float*)d_in[2];
    const float* w1c = (const float*)d_in[3];
    const float* w2a = (const float*)d_in[4];
    const float* w2b = (const float*)d_in[5];
    const float* w2c = (const float*)d_in[6];
    float* out = (float*)d_out;

    char* ws = (char*)d_ws;
    // workspace layout (256-aligned)
    size_t ampOff  = 0;                                  // 2064 f32
    size_t topOff  = 8448;                               // int[8]
    size_t wrawOff = 8704;                               // 64*3 f32
    size_t wsOff   = 9728;                               // 64*3 f32
    size_t wAllOff = 10752;                              // 2*128*63 f32
    size_t xAOff   = 76288;                              // 33.5M bf16 (64MB)
    size_t cosOff  = xAOff  + (size_t)BNT * KSTEPS * 32 * 16 * 2;  // 16MB
    size_t sinOff  = cosOff + (size_t)FT  * KSTEPS * 32 * 16 * 2;  // 16MB
    size_t partOff = sinOff + (size_t)FT  * KSTEPS * 32 * 16 * 2;  // 4MB

    float* amp   = (float*)(ws + ampOff);
    int*   topP  = (int*)  (ws + topOff);
    float* wraw  = (float*)(ws + wrawOff);
    float* wsoft = (float*)(ws + wsOff);
    float* wAll  = (float*)(ws + wAllOff);
    bf16*  xA    = (bf16*) (ws + xAOff);
    bf16*  cosW  = (bf16*) (ws + cosOff);
    bf16*  sinW  = (bf16*) (ws + sinOff);
    float* part  = (float*)(ws + partOff);

    tpi_init_amp<<<1, 64, 0, stream>>>(amp);
    tpi_merge_w<<<(NN * 63 + 255) / 256, 256, 0, stream>>>(w1a, w1b, w1c,
                                                           w2a, w2b, w2c, wAll);
    tpi_gen_tw<<<(FT * KSTEPS * 32) / 256, 256, 0, stream>>>(cosW, sinW);
    tpi_gen_xa<<<(BNT * KSTEPS * 32) / 256, 256, 0, stream>>>(x, xA);
    tpi_dft<<<(BNT / 2) * (FT / 2) / 8, 256, 0, stream>>>(xA, cosW, sinW, part);
    tpi_reduce_amp<<<FBINS / 256, 256, 0, stream>>>(part, amp);
    tpi_topk<<<1, 256, 0, stream>>>(amp, topP);
    tpi_weights<<<dim3(BB, 3), 128, 0, stream>>>(x, topP, wraw);
    tpi_softmax<<<1, 64, 0, stream>>>(wraw, wsoft);
    tpi_copy_out<<<(BB * TT * NN / 4) / 256, 256, 0, stream>>>(x, out);
    dim3 bgrid(TT / CHUNK, BB, NN / NC);
    for (int kk = 0; kk < 3; ++kk)
        tpi_branch<<<bgrid, 256, 0, stream>>>(x, out, wAll, topP, wsoft, kk);
}
